// Block_55138790146349
// MI455X (gfx1250) — compile-verified, matrix operand order checked by reading, not run on hardware
//
#include <hip/hip_runtime.h>
#include <math.h>

// ---------------------------------------------------------------------------
// Transformer block for MI455X (gfx1250).
// Compute-bound (~450 GFLOP): bf16 V_WMMA_F32_16X16X32_BF16 with f32 accum.
// GEMMs: 128x256 block tile, TDM (tensor_load_to_lds) double-buffered LDS
// staging when available; flash-style causal attention (no T x T logits).
// ---------------------------------------------------------------------------

#define DIMD   2048
#define HEADS  16
#define HDIM   128
#define DFF    8192
#define BATCH  2
#define SEQ    2048
#define MTOT   (BATCH * SEQ)

typedef __attribute__((ext_vector_type(16))) __bf16 v16bf;
typedef __attribute__((ext_vector_type(8)))  float  v8f;
typedef __attribute__((ext_vector_type(4)))  unsigned int u32x4;
typedef __attribute__((ext_vector_type(4)))  int i32x4;
typedef __attribute__((ext_vector_type(8)))  int i32x8;

#if defined(__AMDGCN__)
#  if defined(__has_builtin)
#    if __has_builtin(__builtin_amdgcn_tensor_load_to_lds)
#      if __has_builtin(__builtin_amdgcn_s_wait_tensorcnt)
#        define USE_TDM 1
#      endif
#    endif
#  endif
#endif
#ifndef USE_TDM
#define USE_TDM 0
#endif

union Frag {
    v16bf          bf;
    u32x4          q[2];
    unsigned short h[16];
};

__device__ __forceinline__ unsigned short f2bf(float f) {
    union { float f; unsigned u; } x; x.f = f;
    unsigned r = (x.u + 0x7FFFu + ((x.u >> 16) & 1u)) >> 16;   // RNE
    return (unsigned short)r;
}

__device__ __forceinline__ v8f v8zero() {
    v8f v = {0.f, 0.f, 0.f, 0.f, 0.f, 0.f, 0.f, 0.f};
    return v;
}

#if USE_TDM
// ---------------------------------------------------------------------------
// Tensor Data Mover: DMA a [tileRows x 32] bf16 tile (row stride K elems)
// into LDS with 16B padding every 64B row -> LDS row stride 40 elems.
// 2-D descriptor: data_size=2B, tile_dim0=32, pad_interval=16DW, pad=4DW.
// ---------------------------------------------------------------------------
__device__ __forceinline__ void tdm_load_tile(const unsigned short* lds,
                                              const unsigned short* gbase,
                                              int row0, int tileRows,
                                              int K, int rowsTotal, int k0) {
    unsigned lds_off = (unsigned)(unsigned long long)lds;      // LDS byte offset
    unsigned long long ga =
        (unsigned long long)(const void*)(gbase + (size_t)row0 * K + k0);

    u32x4 g0 = { 1u,                                           // count=1 (valid)
                 lds_off,
                 (unsigned)ga,
                 ((unsigned)((ga >> 32) & 0x1FFFFFFull)) | (2u << 30) }; // type=2

    // group1 bitfields (see CDNA5 ISA)
    unsigned long long d0 = ((unsigned long long)((1u << 16)   // data_size=2B
                                                 | (1u << 20)  // pad_enable
                                                 | (3u << 22)  // pad_interval=16DW
                                                 | (3u << 25)))// pad_amount=4DW
                          | ((unsigned long long)((unsigned)K & 0xFFFFu) << 48);
    unsigned long long d1 = ((unsigned long long)((unsigned)K >> 16))
                          | ((unsigned long long)(unsigned)rowsTotal << 16)
                          | (32ull << 48);                     // tile_dim0=32
    unsigned long long d2 = (unsigned long long)(unsigned)tileRows // tile_dim1
                          | ((unsigned long long)(unsigned)K << 32); // stride0 lo
    unsigned long long d3 = 0;                                  // stride0 hi, etc.
    i32x8 g1 = { (int)d0, (int)(d0 >> 32), (int)d1, (int)(d1 >> 32),
                 (int)d2, (int)(d2 >> 32), (int)d3, (int)(d3 >> 32) };
    i32x4 gz = { 0, 0, 0, 0 };
#if __clang_major__ >= 23
    i32x8 gz8 = { 0, 0, 0, 0, 0, 0, 0, 0 };
    __builtin_amdgcn_tensor_load_to_lds(g0, g1, gz, gz, gz8, 0);
#else
    __builtin_amdgcn_tensor_load_to_lds(g0, g1, gz, gz, 0);
#endif
}
#endif

// ---------------------------------------------------------------------------
// fp32 -> bf16 elementwise (weight conversion)
// ---------------------------------------------------------------------------
__global__ void f32_to_bf16_kernel(const float* __restrict__ S,
                                   unsigned short* __restrict__ D, size_t n) {
    size_t i      = (size_t)blockIdx.x * blockDim.x + threadIdx.x;
    size_t stride = (size_t)gridDim.x * blockDim.x;
    for (; i < n; i += stride) D[i] = f2bf(S[i]);
}

// ---------------------------------------------------------------------------
// RMSNorm over rows of length DIMD, bf16 output.  1 block / row, 8 waves.
// ---------------------------------------------------------------------------
__global__ __launch_bounds__(256)
void rmsnorm_bf16_kernel(const float* __restrict__ X,
                         const float* __restrict__ Wg,
                         unsigned short* __restrict__ Y) {
    const int row = blockIdx.x;
    const float* xr = X + (size_t)row * DIMD;

    float s = 0.f;
    for (int i = threadIdx.x; i < DIMD; i += 256) { float v = xr[i]; s += v * v; }
    for (int o = 16; o > 0; o >>= 1) s += __shfl_xor(s, o, 32);

    __shared__ float red[8];
    if ((threadIdx.x & 31) == 0) red[threadIdx.x >> 5] = s;
    __syncthreads();
    float tot = 0.f;
    for (int w = 0; w < 8; ++w) tot += red[w];

    const float rinv = rsqrtf(tot * (1.0f / DIMD) + 1e-5f);
    for (int i = threadIdx.x; i < DIMD; i += 256)
        Y[(size_t)row * DIMD + i] = f2bf(xr[i] * rinv * Wg[i]);
}

// ---------------------------------------------------------------------------
// bf16 NT GEMM:  C[M,N] = A[M,K] * W[N,K]^T  (both K-contiguous, f32 accum)
// Block tile 128(M) x 256(N), BK=32, 8 waves each 64x64 (4x4 WMMA tiles).
// Double-buffered LDS; staging via TDM (async, wave 0 issues, TENSORcnt)
// or manual register double-buffer fallback.
// EPI: 0 -> bf16 store, 1 -> GELU->bf16, 2 -> f32 (+ residual Res) store.
// ---------------------------------------------------------------------------
template <int EPI>
__global__ __launch_bounds__(256)
void gemm_bf16_nt(const unsigned short* __restrict__ A,
                  const unsigned short* __restrict__ W,
                  void* __restrict__ Cout,
                  const float* __restrict__ Res,
                  int M, int N, int K) {
    __shared__ __align__(16) unsigned short sA[2][128][40];
    __shared__ __align__(16) unsigned short sB[2][256][40];

    const int tid  = threadIdx.x;
    const int lane = tid & 31, wave = tid >> 5;
    const int wn = wave & 3, wm = wave >> 2;          // 2 (M) x 4 (N) wave grid
    const int mBase = blockIdx.y * 128;
    const int nBase = blockIdx.x * 256;
    const int mW = wm * 64;                           // wave tile origin in block
    const int nW = wn * 64;
    const int lr   = lane & 15;
    const int half = lane >> 4;
    const int koff = half * 8;                        // ISA 16-bit frag layout

    v8f acc[4][4];
    for (int i = 0; i < 4; ++i)
        for (int j = 0; j < 4; ++j) acc[i][j] = v8zero();

    const int nt = K / 32;

#if USE_TDM
    if (wave == 0) {
        tdm_load_tile(&sA[0][0][0], A, mBase, 128, K, M, 0);
        tdm_load_tile(&sB[0][0][0], W, nBase, 256, K, N, 0);
    }
#else
    for (int it = 0; it < 2; ++it) {                  // prologue: fill buf 0
        int idx = tid + it * 256, row = idx >> 2, ch = (idx & 3) * 8;
        *(u32x4*)&sA[0][row][ch] = *(const u32x4*)(A + (size_t)(mBase + row) * K + ch);
    }
    for (int it = 0; it < 4; ++it) {
        int idx = tid + it * 256, row = idx >> 2, ch = (idx & 3) * 8;
        *(u32x4*)&sB[0][row][ch] = *(const u32x4*)(W + (size_t)(nBase + row) * K + ch);
    }
#endif

    for (int kt = 0; kt < nt; ++kt) {
#if USE_TDM
        if (wave == 0) __builtin_amdgcn_s_wait_tensorcnt(0);
        __syncthreads();
        if (wave == 0 && kt + 1 < nt) {               // async DMA of next tiles
            tdm_load_tile(&sA[(kt + 1) & 1][0][0], A, mBase, 128, K, M, (kt + 1) * 32);
            tdm_load_tile(&sB[(kt + 1) & 1][0][0], W, nBase, 256, K, N, (kt + 1) * 32);
        }
#else
        __syncthreads();
        u32x4 pa[2], pb[4];
        if (kt + 1 < nt) {                            // global loads overlap compute
            int k0n = (kt + 1) * 32;
            for (int it = 0; it < 2; ++it) {
                int idx = tid + it * 256, row = idx >> 2, ch = (idx & 3) * 8;
                pa[it] = *(const u32x4*)(A + (size_t)(mBase + row) * K + k0n + ch);
            }
            for (int it = 0; it < 4; ++it) {
                int idx = tid + it * 256, row = idx >> 2, ch = (idx & 3) * 8;
                pb[it] = *(const u32x4*)(W + (size_t)(nBase + row) * K + k0n + ch);
            }
        }
#endif
        const unsigned short (*tA)[40] = sA[kt & 1];
        const unsigned short (*tB)[40] = sB[kt & 1];

        Frag bfr[4];                                  // all B frags resident
        for (int j = 0; j < 4; ++j) {
            int r = nW + j * 16 + lr;
            bfr[j].q[0] = *(const u32x4*)&tB[r][koff];
            bfr[j].q[1] = *(const u32x4*)&tB[r][koff + 16];
        }
        Frag afr[2];                                  // A frags double-buffered
        afr[0].q[0] = *(const u32x4*)&tA[mW + lr][koff];
        afr[0].q[1] = *(const u32x4*)&tA[mW + lr][koff + 16];
        for (int i = 0; i < 4; ++i) {
            if (i < 3) {
                int r = mW + (i + 1) * 16 + lr;
                afr[(i + 1) & 1].q[0] = *(const u32x4*)&tA[r][koff];
                afr[(i + 1) & 1].q[1] = *(const u32x4*)&tA[r][koff + 16];
            }
            for (int j = 0; j < 4; ++j)
                acc[i][j] = __builtin_amdgcn_wmma_f32_16x16x32_bf16(
                    false, afr[i & 1].bf, false, bfr[j].bf, (short)0, acc[i][j],
                    false, false);
        }
#if !USE_TDM
        if (kt + 1 < nt) {                            // commit next buffer
            int nb = (kt + 1) & 1;
            for (int it = 0; it < 2; ++it) {
                int idx = tid + it * 256, row = idx >> 2, ch = (idx & 3) * 8;
                *(u32x4*)&sA[nb][row][ch] = pa[it];
            }
            for (int it = 0; it < 4; ++it) {
                int idx = tid + it * 256, row = idx >> 2, ch = (idx & 3) * 8;
                *(u32x4*)&sB[nb][row][ch] = pb[it];
            }
        }
#endif
    }

    // epilogue (C layout: lane -> N = lane%16, VGPR r -> M = r + 8*(lane/16))
    for (int i = 0; i < 4; ++i)
        for (int j = 0; j < 4; ++j)
            for (int r = 0; r < 8; ++r) {
                int m = mBase + mW + i * 16 + r + 8 * half;
                int n = nBase + nW + j * 16 + lr;
                size_t o = (size_t)m * N + n;
                float c  = acc[i][j][r];
                if (EPI == 0) {
                    ((unsigned short*)Cout)[o] = f2bf(c);
                } else if (EPI == 1) {
                    float g = 0.5f * c * (1.0f + erff(c * 0.70710678118654752f));
                    ((unsigned short*)Cout)[o] = f2bf(g);
                } else {
                    ((float*)Cout)[o] = c + Res[o];
                }
            }
}

// ---------------------------------------------------------------------------
// Flash-style causal attention.  Grid (SEQ/128, HEADS, BATCH), 256 threads.
// Each of 8 waves owns 16 q-rows; 64-row K/V tiles staged once per 128 q-rows.
// V transposed into LDS so P@V B-fragments are contiguous b128 LDS reads.
// Online softmax in f32 C-fragments with 16-lane shuffle reductions.
// ---------------------------------------------------------------------------
__global__ __launch_bounds__(256)
void flash_attn_kernel(const unsigned short* __restrict__ Q,
                       const unsigned short* __restrict__ Km,
                       const unsigned short* __restrict__ Vm,
                       unsigned short* __restrict__ O) {
    __shared__ __align__(16) unsigned short sK[64][136];   // 64 kv x 128 d (+pad)
    __shared__ __align__(16) unsigned short sVt[128][72];  // 128 d x 64 kv (+pad)
    __shared__ __align__(16) unsigned short sP[8][16][72]; // per-wave P tile

    const int tid  = threadIdx.x;
    const int lane = tid & 31, wave = tid >> 5;
    const int lr   = lane & 15, half = lane >> 4;
    const int koff = half * 8;
    const int qb = blockIdx.x * 128;
    const int h  = blockIdx.y;
    const int b  = blockIdx.z;
    const size_t rowOff  = (size_t)b * SEQ;
    const int    colBase = h * HDIM;
    const int    qw = qb + wave * 16;

    Frag aq[4];                                   // Q: 16 rows x 128 d
    {
        const unsigned short* qp = Q + (rowOff + qw + lr) * DIMD + colBase;
        for (int kc = 0; kc < 4; ++kc) {
            aq[kc].q[0] = *(const u32x4*)(qp + kc * 32 + koff);
            aq[kc].q[1] = *(const u32x4*)(qp + kc * 32 + koff + 16);
        }
    }

    v8f o_acc[8];
    for (int f = 0; f < 8; ++f) o_acc[f] = v8zero();
    float mrun[8], lrun[8];
    for (int r = 0; r < 8; ++r) { mrun[r] = -1e30f; lrun[r] = 0.f; }
    const float scale = 0.08838834764831845f;     // 1/sqrt(128)
    const int kbEnd = qb + 112;                   // diag block of wave 7

    for (int kb = 0; kb <= kbEnd; kb += 64) {
        __syncthreads();
        for (int idx = tid; idx < 64 * 16; idx += 256) {
            int row = idx >> 4, ch = (idx & 15) * 8;
            *(u32x4*)&sK[row][ch] =
                *(const u32x4*)(Km + (rowOff + kb + row) * DIMD + colBase + ch);
            Frag v;
            v.q[0] = *(const u32x4*)(Vm + (rowOff + kb + row) * DIMD + colBase + ch);
            for (int e = 0; e < 8; ++e) sVt[ch + e][row] = v.h[e];   // transpose
        }
        __syncthreads();
        if (kb + 64 <= kbEnd) {                   // hint next KV block into cache
            __builtin_prefetch(Km + (rowOff + kb + 64 + (tid & 63)) * DIMD + colBase, 0, 3);
            __builtin_prefetch(Vm + (rowOff + kb + 64 + (tid & 63)) * DIMD + colBase, 0, 3);
        }

        // S = Q K^T : 16 x 64 per wave (B-frags double-buffered)
        v8f s[4];
        for (int j = 0; j < 4; ++j) s[j] = v8zero();
        Frag bk[2];
        bk[0].q[0] = *(const u32x4*)&sK[lr][koff];
        bk[0].q[1] = *(const u32x4*)&sK[lr][koff + 16];
        for (int t = 0; t < 16; ++t) {
            int j = t >> 2, kc = t & 3;
            if (t + 1 < 16) {
                int jn = (t + 1) >> 2, kn = (t + 1) & 3;
                int r = jn * 16 + lr;
                bk[(t + 1) & 1].q[0] = *(const u32x4*)&sK[r][kn * 32 + koff];
                bk[(t + 1) & 1].q[1] = *(const u32x4*)&sK[r][kn * 32 + koff + 16];
            }
            s[j] = __builtin_amdgcn_wmma_f32_16x16x32_bf16(
                false, aq[kc].bf, false, bk[t & 1].bf, (short)0, s[j], false, false);
        }

        // scale + causal mask (data select, no divergence around WMMA)
        for (int j = 0; j < 4; ++j)
            for (int r = 0; r < 8; ++r) {
                int tq = qw + r + 8 * half;
                int tk = kb + j * 16 + lr;
                float v = s[j][r] * scale;
                s[j][r] = (tk > tq) ? -1e30f : v;
            }

        // online softmax
        float corr[8];
        for (int r = 0; r < 8; ++r) {
            float mx = fmaxf(fmaxf(s[0][r], s[1][r]), fmaxf(s[2][r], s[3][r]));
            for (int d = 8; d > 0; d >>= 1) mx = fmaxf(mx, __shfl_xor(mx, d, 32));
            float mnew = fmaxf(mrun[r], mx);
            corr[r] = __expf(mrun[r] - mnew);
            mrun[r] = mnew;
        }
        float rs[8];
        for (int r = 0; r < 8; ++r) rs[r] = 0.f;
        for (int j = 0; j < 4; ++j)
            for (int r = 0; r < 8; ++r) {
                float p = __expf(s[j][r] - mrun[r]);
                sP[wave][r + 8 * half][j * 16 + lr] = f2bf(p);
                rs[r] += p;
            }
        for (int r = 0; r < 8; ++r) {
            for (int d = 8; d > 0; d >>= 1) rs[r] += __shfl_xor(rs[r], d, 32);
            lrun[r] = lrun[r] * corr[r] + rs[r];
        }
        for (int f = 0; f < 8; ++f)
            for (int r = 0; r < 8; ++r) o_acc[f][r] *= corr[r];

        // O += P V  (P via per-wave LDS roundtrip; V^T rows contiguous)
        for (int kc = 0; kc < 2; ++kc) {
            Frag ap;
            ap.q[0] = *(const u32x4*)&sP[wave][lr][kc * 32 + koff];
            ap.q[1] = *(const u32x4*)&sP[wave][lr][kc * 32 + koff + 16];
            Frag bv[2];
            bv[0].q[0] = *(const u32x4*)&sVt[lr][kc * 32 + koff];
            bv[0].q[1] = *(const u32x4*)&sVt[lr][kc * 32 + koff + 16];
            for (int f = 0; f < 8; ++f) {
                if (f < 7) {
                    int dr = (f + 1) * 16 + lr;
                    bv[(f + 1) & 1].q[0] = *(const u32x4*)&sVt[dr][kc * 32 + koff];
                    bv[(f + 1) & 1].q[1] = *(const u32x4*)&sVt[dr][kc * 32 + koff + 16];
                }
                o_acc[f] = __builtin_amdgcn_wmma_f32_16x16x32_bf16(
                    false, ap.bf, false, bv[f & 1].bf, (short)0, o_acc[f], false, false);
            }
        }
    }

    for (int f = 0; f < 8; ++f)
        for (int r = 0; r < 8; ++r) {
            int tq = qw + r + 8 * half;
            O[(rowOff + tq) * DIMD + colBase + f * 16 + lr] =
                f2bf(o_acc[f][r] / lrun[r]);
        }
}

// ---------------------------------------------------------------------------
// Orchestration
// ---------------------------------------------------------------------------
extern "C" void kernel_launch(void* const* d_in, const int* in_sizes, int n_in,
                              void* d_out, int out_size, void* d_ws, size_t ws_size,
                              hipStream_t stream) {
    (void)in_sizes; (void)n_in; (void)out_size; (void)ws_size;
    const float* x   = (const float*)d_in[0];
    const float* anw = (const float*)d_in[1];
    const float* wq  = (const float*)d_in[2];
    const float* wk  = (const float*)d_in[3];
    const float* wv  = (const float*)d_in[4];
    const float* wo  = (const float*)d_in[5];
    const float* fnw = (const float*)d_in[6];
    const float* w1  = (const float*)d_in[7];
    const float* w2  = (const float*)d_in[8];
    float* out = (float*)d_out;

    const size_t DD = (size_t)DIMD * DIMD;
    const size_t FD = (size_t)DFF * DIMD;
    const size_t MD = (size_t)MTOT * DIMD;

    char*  ws  = (char*)d_ws;
    size_t off = 0;
    auto carve = [&](size_t elems) -> unsigned short* {
        unsigned short* p = (unsigned short*)(ws + off);
        off += ((elems * 2 + 255) & ~(size_t)255);
        return p;
    };
    unsigned short* wqb = carve(DD);
    unsigned short* wkb = carve(DD);
    unsigned short* wvb = carve(DD);
    unsigned short* wob = carve(DD);
    unsigned short* w1b = carve(FD);
    unsigned short* w2b = carve(FD);
    unsigned short* xnb = carve(MD);
    unsigned short* qbf = carve(MD);
    unsigned short* kbf = carve(MD);
    unsigned short* vbf = carve(MD);
    unsigned short* aob = carve(MD);
    unsigned short* hbf = xnb;   // alias: xn dead after QKV GEMMs
    unsigned short* gbf = qbf;   // alias: q..ao (4*MD elems = 67MB) dead after wo

    // 1) weights -> bf16 (100MB total -> resident in 192MB L2)
    f32_to_bf16_kernel<<<1024, 256, 0, stream>>>(wq, wqb, DD);
    f32_to_bf16_kernel<<<1024, 256, 0, stream>>>(wk, wkb, DD);
    f32_to_bf16_kernel<<<1024, 256, 0, stream>>>(wv, wvb, DD);
    f32_to_bf16_kernel<<<1024, 256, 0, stream>>>(wo, wob, DD);
    f32_to_bf16_kernel<<<1024, 256, 0, stream>>>(w1, w1b, FD);
    f32_to_bf16_kernel<<<1024, 256, 0, stream>>>(w2, w2b, FD);

    // 2) attn rmsnorm -> bf16
    rmsnorm_bf16_kernel<<<MTOT, 256, 0, stream>>>(x, anw, xnb);

    // 3) Q, K, V projections (bf16 out)
    dim3 g1(DIMD / 256, MTOT / 128);
    gemm_bf16_nt<0><<<g1, 256, 0, stream>>>(xnb, wqb, qbf, nullptr, MTOT, DIMD, DIMD);
    gemm_bf16_nt<0><<<g1, 256, 0, stream>>>(xnb, wkb, kbf, nullptr, MTOT, DIMD, DIMD);
    gemm_bf16_nt<0><<<g1, 256, 0, stream>>>(xnb, wvb, vbf, nullptr, MTOT, DIMD, DIMD);

    // 4) causal attention
    flash_attn_kernel<<<dim3(SEQ / 128, HEADS, BATCH), 256, 0, stream>>>(qbf, kbf, vbf, aob);

    // 5) output projection + residual (f32 -> d_out)
    gemm_bf16_nt<2><<<g1, 256, 0, stream>>>(aob, wob, out, x, MTOT, DIMD, DIMD);

    // 6) ff rmsnorm -> bf16
    rmsnorm_bf16_kernel<<<MTOT, 256, 0, stream>>>(out, fnw, hbf);

    // 7) GELU(h @ w1^T) -> bf16
    dim3 g2(DFF / 256, MTOT / 128);
    gemm_bf16_nt<1><<<g2, 256, 0, stream>>>(hbf, w1b, gbf, nullptr, MTOT, DFF, DIMD);

    // 8) d_out += g @ w2^T (in-place residual)
    gemm_bf16_nt<2><<<g1, 256, 0, stream>>>(gbf, w2b, out, out, MTOT, DIMD, DFF);
}
